// GATlayer_IMP8_20529943674898
// MI455X (gfx1250) — compile-verified, matrix operand order checked by reading, not run on hardware
//
#include <hip/hip_runtime.h>
#include <hip/hip_bf16.h>

typedef __attribute__((ext_vector_type(16))) _Float16     v16h;
typedef __attribute__((ext_vector_type(8)))  float        v8f;
typedef __attribute__((ext_vector_type(8)))  unsigned int v8u;
typedef __attribute__((ext_vector_type(4)))  unsigned int v4u;
typedef __attribute__((ext_vector_type(8)))  int          v8i_t;
typedef __attribute__((ext_vector_type(4)))  int          v4i_t;

#define GB   4
#define GN   2048
#define GFIN 128
#define GH   8
#define GD   64
#define GHD  512            // H*D
#define BH   (GB*GH)        // 32
#define ROWS (GB*GN)        // 8192
#define OUT0 (ROWS*GHD)     // offset of adj copy inside d_out

#if defined(__has_builtin)
#if __has_builtin(__builtin_amdgcn_tensor_load_to_lds) && __has_builtin(__builtin_amdgcn_s_wait_tensorcnt)
#define GAT_HAS_TDM 1
#endif
#endif
#ifndef GAT_HAS_TDM
#define GAT_HAS_TDM 0
#endif

// K index for slot s (0..15) of a v16h A/B operand, per CDNA5 ISA 16-bit
// 16x32 A-matrix layout: VGPR v=s/2, half p=s%2;
//  lanes 0-15:  V0..V3 -> K 0..7,  V4..V7 -> K 16..23
//  lanes 16-31: V0..V3 -> K 8..15, V4..V7 -> K 24..31
__device__ __forceinline__ int kidx(int s, int hi) {
    int v = s >> 1, p = s & 1;
    return ((v < 4) ? 0 : 16) + (hi ? 8 : 0) + ((v & 3) << 1) + p;
}

__device__ __forceinline__ v8f wmma_f16(v16h a, v16h b, v8f c) {
    return __builtin_amdgcn_wmma_f32_16x16x32_f16(false, a, false, b,
                                                  (short)0, c, false, false);
}

__device__ __forceinline__ float lrelu02(float t) {
    return fmaxf(t, 0.2f * t);       // slope 0.2, branchless
}

// ---------------------------------------------------------------------------
// TDM 2D tile load: Global -> LDS via the Tensor Data Mover.
// Descriptor per CDNA5 ISA, data_size = 8 bytes.
//   row_units     : tile_dim0 / tensor_dim0 (row length, 8B units)
//   rows          : tile_dim1 / tensor_dim1
//   stride_units  : tensor_dim0_stride (8B units)
// Group2/3 zero (2D). Tracked on TENSORcnt (per-wave -> no block barrier).
// This toolchain's builtin is the 6-arg form:
//   (v4u g0, v8i g1, v4i g2, v4i g3, v8i pad, i32 cpol)
// ---------------------------------------------------------------------------
__device__ __forceinline__ void tdm_load_2d(uint32_t lds_addr, const void* gptr,
                                            int row_units, int rows,
                                            int stride_units) {
#if GAT_HAS_TDM
    const uint64_t ga = (uint64_t)gptr;
    v4u g0;
    g0.x = 1u;                                                  // count=1 (valid)
    g0.y = (unsigned)__builtin_amdgcn_readfirstlane((int)lds_addr);
    g0.z = (unsigned)__builtin_amdgcn_readfirstlane((int)(uint32_t)ga);
    g0.w = (unsigned)__builtin_amdgcn_readfirstlane(
              (int)(((uint32_t)(ga >> 32) & 0x01FFFFFFu) | (2u << 30))); // type=2
    v8i_t g1;
    g1[0] = (3 << 16);            // data_size = 8B; no multicast/pad/iterate
    g1[1] = row_units << 16;      // tensor_dim0[15:0]
    g1[2] = rows << 16;           // tensor_dim1[15:0]
    g1[3] = row_units << 16;      // tile_dim0
    g1[4] = rows;                 // tile_dim1 (tile_dim2 = 0)
    g1[5] = stride_units;         // tensor_dim0_stride[31:0]
    g1[6] = 0;                    // stride hi / tensor_dim1_stride lo
    g1[7] = 0;
    const v4i_t z4 = {0, 0, 0, 0};
    const v8i_t z8 = {0, 0, 0, 0, 0, 0, 0, 0};
    __builtin_amdgcn_tensor_load_to_lds(g0, g1, z4, z4, z8, 0);
#endif
}

// ---------------------------------------------------------------------------
// Kernel 1: proj = x@W_proj (f16, TRANSPOSED [bh][d][n] to ws),
//           skip = x@W_skip + bias (f32 into d_out).
// One wave per 16x16 output tile, K=128 as 4 x wmma_f32_16x16x32_f16 per
// weight matrix; the x A-operand (float2 pair loads) is reused for both.
// ---------------------------------------------------------------------------
__global__ __launch_bounds__(256) void gat_proj_kernel(
    const float* __restrict__ x, const float* __restrict__ Wp,
    const float* __restrict__ Ws, const float* __restrict__ bias,
    _Float16* __restrict__ proj_h, float* __restrict__ out_skip) {

    const int lane = threadIdx.x & 31;
    const int wave = threadIdx.x >> 5;
    const int task = blockIdx.x * 8 + wave;          // 0..16383
    const int mtile = task >> 5;                     // 0..511
    const int ntile = task & 31;                     // 0..31
    const int ln = lane & 15;
    const int hi = lane >> 4;

    const int row = mtile * 16 + ln;                 // A row (M = lane%16)
    const int col = ntile * 16 + ln;                 // B/C col (N = lane%16)

    v8f accP = {}; v8f accS = {};
    #pragma unroll
    for (int kt = 0; kt < 4; ++kt) {                 // K = 128 in steps of 32
        v16h a, bp, bs;
        #pragma unroll
        for (int v = 0; v < 8; ++v) {                // K slot pairs are contiguous
            const int k = kt * 32 + kidx(2 * v, hi);
            const float2 xa = *(const float2*)&x[row * GFIN + k];
            a[2 * v]     = (_Float16)xa.x;
            a[2 * v + 1] = (_Float16)xa.y;
        }
        #pragma unroll
        for (int s = 0; s < 16; ++s) {
            const int k = kt * 32 + kidx(s, hi);
            bp[s] = (_Float16)Wp[k * GHD + col];
            bs[s] = (_Float16)Ws[k * GHD + col];
        }
        accP = wmma_f16(a, bp, accP);
        accS = wmma_f16(a, bs, accS);
    }

    const float bv = bias[col];
    const int h = col >> 6, d = col & 63;
    #pragma unroll
    for (int r = 0; r < 8; ++r) {
        const int grow = mtile * 16 + r + hi * 8;    // C row M = r + 8*(lane>=16)
        const int b = grow >> 11, n = grow & 2047;
        proj_h[(((size_t)(b * GH + h) * GD + d) * GN) + n] = (_Float16)accP[r];
        out_skip[(size_t)grow * GHD + col] = accS[r] + bv;  // consumed by kernel 4
    }
}

// ---------------------------------------------------------------------------
// Kernel 2: s_src[bh,n] = proj[bh,:,n].a_src[h], s_tgt likewise. 64K threads.
// proj is [bh][d][n] -> fixed d, lane-consecutive n: fully coalesced.
// ---------------------------------------------------------------------------
__global__ __launch_bounds__(256) void gat_scores_kernel(
    const _Float16* __restrict__ proj_h, const float* __restrict__ a_src,
    const float* __restrict__ a_tgt, float* __restrict__ s_src,
    float* __restrict__ s_tgt) {

    const int idx = blockIdx.x * 256 + threadIdx.x;  // 0..65535 = bh*2048+n
    const int bh = idx >> 11, n = idx & 2047, h = bh & 7;
    const _Float16* p = proj_h + (size_t)bh * GD * GN + n;
    float ss = 0.f, st = 0.f;
    #pragma unroll
    for (int d = 0; d < GD; ++d) {
        const float pv = (float)p[(size_t)d * GN];
        ss += pv * a_src[h * GD + d];
        st += pv * a_tgt[h * GD + d];
    }
    s_src[idx] = ss;
    s_tgt[idx] = st;
}

// ---------------------------------------------------------------------------
// Kernel 3: safe-softmax pass 1 — per (bh,i) row: m = max_j score, l = sum exp.
// One wave32 per row, lanes stride j, butterfly (m,l) reduction.
// ---------------------------------------------------------------------------
__global__ __launch_bounds__(256) void gat_softmax_stats_kernel(
    const float* __restrict__ s_src, const float* __restrict__ s_tgt,
    const float* __restrict__ adj, float* __restrict__ m_out,
    float* __restrict__ invl_out) {

    const int lane = threadIdx.x & 31;
    const int ridx = blockIdx.x * 8 + (threadIdx.x >> 5); // 0..65535
    const int bh = ridx >> 11, i = ridx & 2047;
    const float si = s_src[ridx];

    float m = -3.0e38f, l = 0.f;
    for (int j = lane; j < GN; j += 32) {
        float t = lrelu02(si + s_tgt[bh * GN + j]) + adj[(size_t)i * GN + j];
        const float nm = fmaxf(m, t);
        l = l * __expf(m - nm) + __expf(t - nm);
        m = nm;
    }
    #pragma unroll
    for (int off = 16; off > 0; off >>= 1) {
        const float om = __shfl_xor(m, off, 32);
        const float ol = __shfl_xor(l, off, 32);
        const float nm = fmaxf(m, om);
        l = l * __expf(m - nm) + ol * __expf(om - nm);
        m = nm;
    }
    if (lane == 0) { m_out[ridx] = m; invl_out[ridx] = 1.0f / l; }
}

// ---------------------------------------------------------------------------
// Kernel 4: pass 2 — normalized probs -> WMMA AV matmul -> +skip -> ELU.
// One wave per (bh, 16-row i-tile): 4096 wave-tasks, 4 waves/block.
// Per 32-wide j-tile the wave TDM-DMAs the adj tile (16x32 f32) and the
// transposed proj tile (64 d-rows x 32 j f16) into its own double-buffered
// LDS slices; TENSORcnt overlaps tile j+1 DMA with tile j WMMA compute.
// ---------------------------------------------------------------------------
__global__ __launch_bounds__(128) void gat_attn_av_kernel(
    const _Float16* __restrict__ proj_h, const float* __restrict__ adj,
    const float* __restrict__ s_src, const float* __restrict__ s_tgt,
    const float* __restrict__ m_in, const float* __restrict__ invl_in,
    float* __restrict__ out) {

    __shared__ float    adj_t[4][2][16 * 32];     // 16 KB
    __shared__ uint32_t proj_t[4][2][64 * 16];    // 32 KB: [d][j] halves as u32

    const int lane = threadIdx.x & 31;
    const int wave = threadIdx.x >> 5;
    const int task = blockIdx.x * 4 + wave;       // 0..4095
    const int itile = task & 127;
    const int bh = task >> 7;
    const int b = bh >> 3, h = bh & 7;
    const int ln = lane & 15;
    const int hi = lane >> 4;
    const int i0 = itile * 16;

    // per-lane row scalars (A-matrix row M = lane%16)
    const int ridx = bh * GN + i0 + ln;
    const float si   = s_src[ridx];
    const float mi   = m_in[ridx];
    const float invl = invl_in[ridx];

    const _Float16* pbase = proj_h + (size_t)bh * GD * GN;   // [64][2048] f16

    v8f acc0 = {}, acc1 = {}, acc2 = {}, acc3 = {};

#if GAT_HAS_TDM
    // issue both tile DMAs for j-tile jt (wrapped) into buffer `buf`
    auto issue = [&](int jt, int buf) {
        const int j0 = (jt & 127) * 32;
        // adj tile: 16 rows x 128B, row stride 8192B (= 1024 x 8B)
        tdm_load_2d((uint32_t)(uintptr_t)&adj_t[wave][buf][0],
                    adj + (size_t)i0 * GN + j0, 16, 16, GN / 2);
        // proj tile: 64 d-rows x 64B (32 halves), row stride 4096B (= 512 x 8B)
        tdm_load_2d((uint32_t)(uintptr_t)&proj_t[wave][buf][0],
                    pbase + j0, 8, 64, (GN * 2) / 8);
    };
    issue(0, 0);
#endif

    for (int jt = 0; jt < GN / 32; ++jt) {
        const int buf = jt & 1;
        const int j0 = jt * 32;
#if GAT_HAS_TDM
        issue(jt + 1, buf ^ 1);                      // prefetch next tile
        __builtin_amdgcn_s_wait_tensorcnt(2);        // tile jt's 2 DMAs done
#else
        for (int idx = lane; idx < 16 * 32; idx += 32)
            adj_t[wave][buf][idx] =
                adj[(size_t)(i0 + (idx >> 5)) * GN + j0 + (idx & 31)];
        {
            const uint32_t* pv32 = (const uint32_t*)(pbase + j0);
            for (int idx = lane; idx < 64 * 16; idx += 32)
                proj_t[wave][buf][idx] = pv32[(size_t)(idx >> 4) * (GN / 2) + (idx & 15)];
        }
        __syncthreads();
#endif
        // normalized probabilities straight into the A operand layout
        v16h a;
        #pragma unroll
        for (int s = 0; s < 16; ++s) {
            const int k = kidx(s, hi);               // j within tile
            float t = lrelu02(si + s_tgt[bh * GN + j0 + k]) +
                      adj_t[wave][buf][ln * 32 + k];
            a[s] = (_Float16)(__expf(t - mi) * invl);
        }
        // gather B operands (32x16 per 16-wide D block): u32 = 2 consecutive K halves
        const uint32_t* pt = &proj_t[wave][buf][0];
        v8u u0, u1, u2, u3;
        #pragma unroll
        for (int v = 0; v < 8; ++v) {
            const int k2 = kidx(2 * v, hi) >> 1;     // dword index along j
            u0[v] = pt[( 0 + ln) * 16 + k2];
            u1[v] = pt[(16 + ln) * 16 + k2];
            u2[v] = pt[(32 + ln) * 16 + k2];
            u3[v] = pt[(48 + ln) * 16 + k2];
        }
        acc0 = wmma_f16(a, __builtin_bit_cast(v16h, u0), acc0);
        acc1 = wmma_f16(a, __builtin_bit_cast(v16h, u1), acc1);
        acc2 = wmma_f16(a, __builtin_bit_cast(v16h, u2), acc2);
        acc3 = wmma_f16(a, __builtin_bit_cast(v16h, u3), acc3);
#if !GAT_HAS_TDM
        __syncthreads();
#endif
    }

    // combine with skip (already in d_out from kernel 1), ELU, store
    #pragma unroll
    for (int r = 0; r < 8; ++r) {
        const int grow = b * GN + i0 + r + hi * 8;   // C row M = r + 8*hi
        const float av[4] = {acc0[r], acc1[r], acc2[r], acc3[r]};
        #pragma unroll
        for (int db = 0; db < 4; ++db) {
            const int col = h * GD + db * 16 + ln;
            const size_t addr = (size_t)grow * GHD + col;
            const float v = av[db] + out[addr];
            out[addr] = v > 0.f ? v : (__expf(v) - 1.0f);
        }
    }
}

// ---------------------------------------------------------------------------
// Kernel 5: copy adj into the second tuple output slot.
// ---------------------------------------------------------------------------
__global__ __launch_bounds__(256) void gat_adjcopy_kernel(
    const float4* __restrict__ adj4, float4* __restrict__ dst4) {
    const int idx = blockIdx.x * 256 + threadIdx.x;  // 0 .. N*N/4-1
    dst4[idx] = adj4[idx];
}

extern "C" void kernel_launch(void* const* d_in, const int* in_sizes, int n_in,
                              void* d_out, int out_size, void* d_ws, size_t ws_size,
                              hipStream_t stream) {
    const float* x      = (const float*)d_in[0];
    const float* adj    = (const float*)d_in[1];
    const float* W_proj = (const float*)d_in[2];
    const float* a_src  = (const float*)d_in[3];
    const float* a_tgt  = (const float*)d_in[4];
    const float* W_skip = (const float*)d_in[5];
    const float* bias   = (const float*)d_in[6];
    float* out = (float*)d_out;

    // workspace layout (9 MB total)
    char* ws = (char*)d_ws;
    _Float16* proj_h = (_Float16*)ws;                       // 8 MB, [bh][d][n]
    float* s_src = (float*)(ws + (size_t)BH * GN * GD * 2); // 256 KB each
    float* s_tgt = s_src + BH * GN;
    float* m_buf = s_tgt + BH * GN;
    float* l_buf = m_buf + BH * GN;

    // 1) projection GEMMs (skip+bias lands in d_out)
    gat_proj_kernel<<<2048, 256, 0, stream>>>(x, W_proj, W_skip, bias, proj_h, out);
    // 2) per-node attention logits
    gat_scores_kernel<<<BH * GN / 256, 256, 0, stream>>>(proj_h, a_src, a_tgt, s_src, s_tgt);
    // 3) softmax row stats (max, 1/sum)
    gat_softmax_stats_kernel<<<BH * GN / 8, 256, 0, stream>>>(s_src, s_tgt, adj, m_buf, l_buf);
    // 4) attention x V + skip + ELU (TDM double-buffered)
    gat_attn_av_kernel<<<BH * (GN / 16) / 4, 128, 0, stream>>>(proj_h, adj, s_src, s_tgt,
                                                               m_buf, l_buf, out);
    // 5) tuple's second output: adj passthrough
    gat_adjcopy_kernel<<<(GN * GN / 4) / 256, 256, 0, stream>>>(
        (const float4*)adj, (float4*)(out + (size_t)OUT0));
}
